// GCN_57037165691305
// MI455X (gfx1250) — matrix operand order repaired and checked by reading, not compiled, not run on hardware
//
#include <hip/hip_runtime.h>
#include <hip/hip_bf16.h>

// GCN layer: out = PReLU(adj @ (seq @ W^T) + bias)
// B=256, N=512, IN=OUT=64. Memory-bound on the 268MB adj stream ->
// fp32 streamed NT, converted in-register to f16, v_wmma_f32_16x16x32_f16,
// f32 accumulate. f16 intermediate kept cache-resident (default TH).

typedef __attribute__((ext_vector_type(16))) _Float16 v16h;
typedef __attribute__((ext_vector_type(8)))  _Float16 v8h;
typedef __attribute__((ext_vector_type(8)))  float    v8f;
typedef __attribute__((ext_vector_type(4)))  float    v4f;

#define GCN_B   256
#define GCN_N   512
#define GCN_FT  64
#define LDS_STR 72   // f16 elements per o-row in LDS (144B: 16B-aligned, bank-spread)

// ---------------------------------------------------------------------------
// Kernel 1: seq_fts = seq @ W^T  (WMMA f16), emitted as fragment-ready f16
// blobs: fts[b][kstep(0..15)][t(0..3)][lane(0..31)][e(0..15)]
// where element e of (kstep,t,lane): o = 16t + (lane&15),
//                                    n = 32*kstep + 16*(lane>>4) + e
// ---------------------------------------------------------------------------
__global__ __launch_bounds__(128) void gcn_fc_wmma(
    const float* __restrict__ seq, const float* __restrict__ W,
    _Float16* __restrict__ fts)
{
    __shared__ _Float16 ldsT[4][64 * LDS_STR];   // per-wave 64 o-rows x 32 n
    const int wave = threadIdx.x >> 5;
    const int lane = threadIdx.x & 31;
    const int row  = lane & 15;
    const int hi   = lane >> 4;

    const int w       = blockIdx.x * 4 + wave;   // 4096 waves total
    const int rowbase = w * 32;                  // 32 seq rows per wave
    const int b       = rowbase >> 9;
    const int kstep   = (rowbase & 511) >> 5;

    // Preload B fragments of W^T: Wf[s][t], K = 32s+16hi+e, N = 16t+row.
    // W is row-major [o][f] -> contiguous 16 floats per lane (hot, default TH).
    v16h Wf[2][4];
#pragma unroll
    for (int s = 0; s < 2; ++s)
#pragma unroll
        for (int t = 0; t < 4; ++t) {
            const float* wp = W + (t * 16 + row) * GCN_FT + s * 32 + hi * 16;
            v4f c0 = *(const v4f*)(wp);
            v4f c1 = *(const v4f*)(wp + 4);
            v4f c2 = *(const v4f*)(wp + 8);
            v4f c3 = *(const v4f*)(wp + 12);
            v16h f;
#pragma unroll
            for (int e = 0; e < 4; ++e) {
                f[e]      = (_Float16)c0[e];
                f[e + 4]  = (_Float16)c1[e];
                f[e + 8]  = (_Float16)c2[e];
                f[e + 12] = (_Float16)c3[e];
            }
            Wf[s][t] = f;
        }

    v8f acc[2][4];
#pragma unroll
    for (int m = 0; m < 2; ++m)
#pragma unroll
        for (int t = 0; t < 4; ++t) { v8f z = {}; acc[m][t] = z; }

#pragma unroll
    for (int m = 0; m < 2; ++m) {
        const float* srow = seq + (size_t)(rowbase + m * 16 + row) * GCN_FT;
#pragma unroll
        for (int s = 0; s < 2; ++s) {
            // A fragment (16x32 f16): f=32s+16hi+e
            // seq is a single-pass stream -> non-temporal loads.
            const float* p0 = srow + s * 32 + hi * 16;
            v4f c0 = __builtin_nontemporal_load((const v4f*)(p0));
            v4f c1 = __builtin_nontemporal_load((const v4f*)(p0 + 4));
            v4f c2 = __builtin_nontemporal_load((const v4f*)(p0 + 8));
            v4f c3 = __builtin_nontemporal_load((const v4f*)(p0 + 12));
            v16h a;
#pragma unroll
            for (int e = 0; e < 4; ++e) {
                a[e]      = (_Float16)c0[e];
                a[e + 4]  = (_Float16)c1[e];
                a[e + 8]  = (_Float16)c2[e];
                a[e + 12] = (_Float16)c3[e];
            }
#pragma unroll
            for (int t = 0; t < 4; ++t)
                acc[m][t] = __builtin_amdgcn_wmma_f32_16x16x32_f16(
                    false, a, false, Wf[s][t], (short)0, acc[m][t], false, false);
        }
    }

    // Scatter D tiles into LDS transposed: ldsT[o*LDS_STR + nloc]
    _Float16* lt = &ldsT[wave][0];
#pragma unroll
    for (int m = 0; m < 2; ++m)
#pragma unroll
        for (int t = 0; t < 4; ++t) {
            const int o = t * 16 + row;           // D: N = lane&15
#pragma unroll
            for (int v = 0; v < 8; ++v) {
                const int nloc = m * 16 + v + 8 * hi;   // D: M = v + 8*hi
                lt[o * LDS_STR + nloc] = (_Float16)acc[m][t][v];
            }
        }
    __syncthreads();

    // Coalesced blob store: 32B contiguous per lane. Keep default (RT) TH so
    // the blobs stay resident in L2 for kernel 2's 32x reuse per batch.
#pragma unroll
    for (int t = 0; t < 4; ++t) {
        const int o = t * 16 + row;
        const _Float16* src = lt + o * LDS_STR + hi * 16;
        v8h lo8 = *(const v8h*)(src);
        v8h hi8 = *(const v8h*)(src + 8);
        _Float16* dst = fts + ((((size_t)b * 16 + kstep) * 4 + t) * 32 + lane) * 16;
        *(v8h*)dst       = lo8;
        *(v8h*)(dst + 8) = hi8;
    }
}

// ---------------------------------------------------------------------------
// Kernel 2: out = PReLU(adj @ seq_fts + bias).  One wave -> 16x64 output tile.
// A = adj tile fp32->f16 in registers (NT stream); B = fragment-ready blobs.
// ---------------------------------------------------------------------------
__global__ __launch_bounds__(128) void gcn_agg_wmma(
    const float* __restrict__ adj, const _Float16* __restrict__ fts,
    const float* __restrict__ bias, const float* __restrict__ alphap,
    float* __restrict__ out)
{
    const int wave = threadIdx.x >> 5;
    const int lane = threadIdx.x & 31;
    const int row  = lane & 15;
    const int hi   = lane >> 4;

    const int b     = blockIdx.x >> 3;                 // 8 blocks per batch
    const int mtile = (blockIdx.x & 7) * 4 + wave;     // 32 m-tiles per batch
    const int mbase = mtile * 16;

    const float alpha = alphap[0];

    const float* arow = adj + ((size_t)b * GCN_N + mbase + row) * GCN_N;
    const _Float16* fb = fts + (size_t)b * (16 * 4 * 32 * 16);

    v8f acc[4];
#pragma unroll
    for (int t = 0; t < 4; ++t) { v8f z = {}; acc[t] = z; }

#pragma unroll 2
    for (int k = 0; k < 16; ++k) {
        // A fragment: 16 rows of adj, K = 32k .. 32k+31 (268MB one-shot -> NT)
        const float* p0 = arow + k * 32 + hi * 16;
        v4f c0 = __builtin_nontemporal_load((const v4f*)(p0));
        v4f c1 = __builtin_nontemporal_load((const v4f*)(p0 + 4));
        v4f c2 = __builtin_nontemporal_load((const v4f*)(p0 + 8));
        v4f c3 = __builtin_nontemporal_load((const v4f*)(p0 + 12));
        v16h a;
#pragma unroll
        for (int e = 0; e < 4; ++e) {
            a[e]      = (_Float16)c0[e];
            a[e + 4]  = (_Float16)c1[e];
            a[e + 8]  = (_Float16)c2[e];
            a[e + 12] = (_Float16)c3[e];
        }
        const _Float16* bk = fb + ((size_t)k * 4 * 32 + lane) * 16;
#pragma unroll
        for (int t = 0; t < 4; ++t) {
            v16h bf = *(const v16h*)(bk + (size_t)t * 32 * 16);  // 32B/lane, L2-hot
            acc[t] = __builtin_amdgcn_wmma_f32_16x16x32_f16(
                false, a, false, bf, (short)0, acc[t], false, false);
        }
    }

    // Epilogue: bias + PReLU, fp32 NT store (write-once, 64B/half-wave).
#pragma unroll
    for (int t = 0; t < 4; ++t) {
        const int o = t * 16 + row;
        const float bv = bias[o];
        float* obase = out + ((size_t)b * GCN_N + mbase) * GCN_FT + o;
#pragma unroll
        for (int v = 0; v < 8; ++v) {
            float x = acc[t][v] + bv;
            x = (x >= 0.0f) ? x : alpha * x;
            __builtin_nontemporal_store(x, obase + (size_t)(v + 8 * hi) * GCN_FT);
        }
    }
}

// ---------------------------------------------------------------------------
extern "C" void kernel_launch(void* const* d_in, const int* in_sizes, int n_in,
                              void* d_out, int out_size, void* d_ws, size_t ws_size,
                              hipStream_t stream) {
    const float* seq   = (const float*)d_in[0];   // [256,512,64]
    const float* adj   = (const float*)d_in[1];   // [256,512,512]
    const float* W     = (const float*)d_in[2];   // [64,64]
    const float* bias  = (const float*)d_in[3];   // [64]
    const float* alpha = (const float*)d_in[4];   // [1]
    float* out = (float*)d_out;
    _Float16* fts = (_Float16*)d_ws;              // 16 MB f16 intermediate

    // Kernel 1: 4096 waves (32 seq rows each) -> 1024 blocks x 128 threads
    gcn_fc_wmma<<<1024, 128, 0, stream>>>(seq, W, fts);
    // Kernel 2: 8192 waves (one 16x64 tile each) -> 2048 blocks x 128 threads
    gcn_agg_wmma<<<2048, 128, 0, stream>>>(adj, fts, bias, alpha, out);
}
